// Time2Vec_16277926051945
// MI455X (gfx1250) — compile-verified, matrix operand order checked by reading, not run on hardware
//
#include <hip/hip_runtime.h>
#include <hip/hip_bf16.h>
#include <stdint.h>

typedef __attribute__((ext_vector_type(16))) __bf16 v16bf;
typedef __attribute__((ext_vector_type(8)))  __bf16 v8bf;
typedef __attribute__((ext_vector_type(4)))  __bf16 v4bf;
typedef __attribute__((ext_vector_type(8)))  float  v8f;

namespace {
constexpr int IN_F  = 1024;   // K and N of the GEMM
constexpr int OUT_F = 2048;
constexpr int BM = 128;       // block tile rows
constexpr int BN = 64;        // block tile cols
constexpr int BK = 32;        // k-step (bf16 WMMA K)
constexpr int LDA = 40;       // padded LDS row stride in bf16 (80B rows, 16B aligned)
constexpr int LDB = 40;
}

// fp32 -> bf16 hi/lo split (hi = RNE, lo = residual): ~16 mantissa bits total
__device__ __forceinline__ void split2(float x, __bf16& h, __bf16& l) {
    h = (__bf16)x;
    l = (__bf16)(x - (float)h);
}

__device__ __forceinline__ v16bf pack16(v8bf lo, v8bf hi) {
    v16bf r;
#pragma unroll
    for (int i = 0; i < 8; ++i) { r[i] = lo[i]; r[i + 8] = hi[i]; }
    return r;
}

__device__ __forceinline__ v8f wmma_bf16(v16bf a, v16bf b, v8f c) {
    return __builtin_amdgcn_wmma_f32_16x16x32_bf16(
        false, a, false, b, (short)0, c, false, false);
}

// CDNA5 async global->LDS copy, 16B per lane (ISA 10, ASYNCcnt tracked).
// lds_off = byte offset within workgroup LDS (low 32 bits of generic pointer).
__device__ __forceinline__ void async_ld16(uint32_t lds_off, const void* g) {
    asm volatile("global_load_async_to_lds_b128 %0, %1, off"
                 :: "v"(lds_off), "v"(g) : "memory");
}
__device__ __forceinline__ void wait_async0() {
    asm volatile("s_wait_asynccnt 0x0" ::: "memory");
}

// ---------------- prep kernels (tiered on ws_size) ----------------

// Split W (fp32 [k][n]) into transposed bf16 planes Wth/Wtl ([n][k]).
__global__ __launch_bounds__(256) void split_w_kernel(
    const float* __restrict__ W, __bf16* __restrict__ Wth, __bf16* __restrict__ Wtl)
{
    int idx = blockIdx.x * 256 + threadIdx.x;     // over 1024*1024
    int k = idx >> 10, n = idx & 1023;
    __bf16 h, l; split2(W[idx], h, l);
    Wth[(size_t)n * IN_F + k] = h;
    Wtl[(size_t)n * IN_F + k] = l;
}

// Split x into bf16 planes Xh/Xl (row-major) AND emit out[:, :1024] = w*x + p.
__global__ __launch_bounds__(256) void split_x_kernel(
    const float* __restrict__ x, __bf16* __restrict__ Xh, __bf16* __restrict__ Xl,
    const float* __restrict__ wS, const float* __restrict__ pS,
    float* __restrict__ out)
{
    const float w0 = wS[0], p0 = pS[0];
    size_t idx = (size_t)blockIdx.x * 256 + threadIdx.x;   // float4 index
    size_t row = idx >> 8;                                 // 256 float4 per row
    int   c4  = (int)(idx & 255);
    float4 v = ((const float4*)x)[idx];
    float vv[4] = {v.x, v.y, v.z, v.w};
    v4bf h4, l4;
#pragma unroll
    for (int j = 0; j < 4; ++j) {
        __bf16 h, l;
        split2(vv[j], h, l);
        h4[j] = h; l4[j] = l;
    }
    *(v4bf*)(Xh + idx * 4) = h4;
    *(v4bf*)(Xl + idx * 4) = l4;
    float4 o = make_float4(w0 * v.x + p0, w0 * v.y + p0, w0 * v.z + p0, w0 * v.w + p0);
    ((float4*)(out + row * OUT_F))[c4] = o;
}

// ---------------- main GEMM + periodic epilogue ----------------

template <bool XPRE, bool WPRE>
__global__ __launch_bounds__(256) void t2v_gemm_kernel(
    const float* __restrict__ x,  const float* __restrict__ W,
    const float* __restrict__ P,
    const float* __restrict__ wS, const float* __restrict__ pS,
    const float* __restrict__ swS, const float* __restrict__ cwS,
    const __bf16* __restrict__ Wth, const __bf16* __restrict__ Wtl,
    const __bf16* __restrict__ Xh,  const __bf16* __restrict__ Xl,
    float* __restrict__ out)
{
    __shared__ __align__(16) __bf16 Ah[BM][LDA];
    __shared__ __align__(16) __bf16 Al[BM][LDA];
    __shared__ __align__(16) __bf16 Bh[BN][LDB];   // transposed: Bh[n][k]
    __shared__ __align__(16) __bf16 Bl[BN][LDB];

    const int t    = threadIdx.x;
    const int lane = t & 31;
    const int wid  = t >> 5;            // 8 waves: 4 (M) x 2 (N)
    const int wm   = (wid >> 1) * 32;
    const int wn   = (wid & 1)  * 32;
    const int r16  = lane & 15;
    const int hl   = lane >> 4;

    const int m0 = blockIdx.y * BM;
    const int n0 = blockIdx.x * BN;

    const float w0 = wS[0], p0 = pS[0];
    const float sw = swS[0], cw = cwS[0];

    v8f acc[2][2];
#pragma unroll
    for (int i = 0; i < 2; ++i)
#pragma unroll
        for (int j = 0; j < 2; ++j)
#pragma unroll
            for (int e = 0; e < 8; ++e) acc[i][j][e] = 0.0f;

    for (int k0 = 0; k0 < IN_F; k0 += BK) {
        float4 av[4];
        if constexpr (!XPRE) {
#pragma unroll
            for (int i = 0; i < 4; ++i) {
                int id = t + i * 256;
                int row = id >> 3, c4 = id & 7;
                av[i] = ((const float4*)(x + (size_t)(m0 + row) * IN_F + k0))[c4];
            }
        }
        float4 bv[2];
        if constexpr (!WPRE) {
#pragma unroll
            for (int i = 0; i < 2; ++i) {
                int id = t + i * 256;
                int row = id >> 4, c4 = id & 15;
                bv[i] = ((const float4*)(W + (size_t)(k0 + row) * IN_F + n0))[c4];
            }
        }

        __syncthreads();   // previous iteration's fragment reads retired

        if constexpr (XPRE) {
            // 8 KB per A plane = 512 x 16B chunks, 2 per thread, pure async DMA
#pragma unroll
            for (int i = 0; i < 2; ++i) {
                int id = t + i * 256;
                int row = id >> 2, c8 = (id & 3) * 8;
                const size_t g = (size_t)(m0 + row) * IN_F + k0 + c8;
                async_ld16((uint32_t)(uintptr_t)&Ah[row][c8], Xh + g);
                async_ld16((uint32_t)(uintptr_t)&Al[row][c8], Xl + g);
            }
        } else {
#pragma unroll
            for (int i = 0; i < 4; ++i) {
                int id = t + i * 256;
                int row = id >> 3, c = (id & 7) * 4;
                float v[4] = {av[i].x, av[i].y, av[i].z, av[i].w};
#pragma unroll
                for (int j = 0; j < 4; ++j) {
                    __bf16 h, l; split2(v[j], h, l);
                    Ah[row][c + j] = h; Al[row][c + j] = l;
                }
            }
            // fused: out[:, :1024] = w*x + p (x tile already in registers)
            if (blockIdx.x == 0) {
#pragma unroll
                for (int i = 0; i < 4; ++i) {
                    int id = t + i * 256;
                    int row = id >> 3, c4 = id & 7;
                    float4 o;
                    o.x = w0 * av[i].x + p0;  o.y = w0 * av[i].y + p0;
                    o.z = w0 * av[i].z + p0;  o.w = w0 * av[i].w + p0;
                    ((float4*)(out + (size_t)(m0 + row) * OUT_F + k0))[c4] = o;
                }
            }
        }

        if constexpr (WPRE) {
            // 4 KB per B plane = 256 x 16B chunks, 1 per thread
            int row = t >> 2, c8 = (t & 3) * 8;
            const size_t g = (size_t)(n0 + row) * IN_F + k0 + c8;
            async_ld16((uint32_t)(uintptr_t)&Bh[row][c8], Wth + g);
            async_ld16((uint32_t)(uintptr_t)&Bl[row][c8], Wtl + g);
        } else {
#pragma unroll
            for (int i = 0; i < 2; ++i) {
                int id = t + i * 256;
                int row = id >> 4, c = (id & 15) * 4;
                float v[4] = {bv[i].x, bv[i].y, bv[i].z, bv[i].w};
#pragma unroll
                for (int j = 0; j < 4; ++j) {
                    __bf16 h, l; split2(v[j], h, l);
                    Bh[c + j][row] = h; Bl[c + j][row] = l;
                }
            }
        }

        if constexpr (XPRE || WPRE) wait_async0();
        __syncthreads();   // LDS tiles (ds + async) visible to all waves

        // fragments: A layout per ISA 7.12.2 (bf16 16x32); B symmetric via transpose
        v16bf aH[2], aL[2], bH[2], bL[2];
#pragma unroll
        for (int tm = 0; tm < 2; ++tm) {
            int ar = wm + tm * 16 + r16;
            aH[tm] = pack16(*(const v8bf*)&Ah[ar][hl * 8], *(const v8bf*)&Ah[ar][16 + hl * 8]);
            aL[tm] = pack16(*(const v8bf*)&Al[ar][hl * 8], *(const v8bf*)&Al[ar][16 + hl * 8]);
        }
#pragma unroll
        for (int tn = 0; tn < 2; ++tn) {
            int bc = wn + tn * 16 + r16;
            bH[tn] = pack16(*(const v8bf*)&Bh[bc][hl * 8], *(const v8bf*)&Bh[bc][16 + hl * 8]);
            bL[tn] = pack16(*(const v8bf*)&Bl[bc][hl * 8], *(const v8bf*)&Bl[bc][16 + hl * 8]);
        }

        // bf16x3: hi*hi + hi*lo + lo*hi, f32 accumulate
#pragma unroll
        for (int tm = 0; tm < 2; ++tm) {
#pragma unroll
            for (int tn = 0; tn < 2; ++tn) {
                acc[tm][tn] = wmma_bf16(aH[tm], bH[tn], acc[tm][tn]);
                acc[tm][tn] = wmma_bf16(aH[tm], bL[tn], acc[tm][tn]);
                acc[tm][tn] = wmma_bf16(aL[tm], bH[tn], acc[tm][tn]);
            }
        }
    }

    // epilogue: z = acc + P[col]; out[:, 1024:] = sw*sin(z) + cw*cos(z)
    // |z| ~ 32 std dev, well inside v_sin/v_cos hardware domain -> fast intrinsics
#pragma unroll
    for (int tm = 0; tm < 2; ++tm) {
#pragma unroll
        for (int tn = 0; tn < 2; ++tn) {
            int col = n0 + wn + tn * 16 + r16;
            float pv = P[col];
#pragma unroll
            for (int e = 0; e < 8; ++e) {
                int row = m0 + wm + tm * 16 + e + 8 * hl;
                float z = acc[tm][tn][e] + pv;
                out[(size_t)row * OUT_F + IN_F + col] = sw * __sinf(z) + cw * __cosf(z);
            }
        }
    }
}

extern "C" void kernel_launch(void* const* d_in, const int* in_sizes, int n_in,
                              void* d_out, int out_size, void* d_ws, size_t ws_size,
                              hipStream_t stream) {
    const float* x  = (const float*)d_in[0];
    const float* W  = (const float*)d_in[1];
    const float* P  = (const float*)d_in[2];
    const float* w  = (const float*)d_in[3];
    const float* p  = (const float*)d_in[4];
    const float* sw = (const float*)d_in[5];
    const float* cw = (const float*)d_in[6];
    float* out = (float*)d_out;

    const int batch = in_sizes[0] / IN_F;            // 65536
    const size_t needW = (size_t)IN_F * IN_F * 2 * sizeof(__bf16);          // 4 MiB
    const size_t needX = (size_t)batch * IN_F * 2 * sizeof(__bf16);         // 256 MiB

    __bf16* Wth = (__bf16*)d_ws;
    __bf16* Wtl = Wth + (size_t)IN_F * IN_F;
    __bf16* Xh  = (__bf16*)((char*)d_ws + needW);
    __bf16* Xl  = Xh + (size_t)batch * IN_F;

    const bool wpre = ws_size >= needW;
    const bool xpre = ws_size >= needW + needX;

    dim3 grid(IN_F / BN, batch / BM);                // (16, 512)

    if (xpre) {
        split_w_kernel<<<(IN_F * IN_F) / 256, 256, 0, stream>>>(W, Wth, Wtl);
        split_x_kernel<<<(unsigned)((size_t)batch * IN_F / 4 / 256), 256, 0, stream>>>(
            x, Xh, Xl, w, p, out);
        t2v_gemm_kernel<true, true><<<grid, 256, 0, stream>>>(
            x, W, P, w, p, sw, cw, Wth, Wtl, Xh, Xl, out);
    } else if (wpre) {
        split_w_kernel<<<(IN_F * IN_F) / 256, 256, 0, stream>>>(W, Wth, Wtl);
        t2v_gemm_kernel<false, true><<<grid, 256, 0, stream>>>(
            x, W, P, w, p, sw, cw, Wth, Wtl, nullptr, nullptr, out);
    } else {
        t2v_gemm_kernel<false, false><<<grid, 256, 0, stream>>>(
            x, W, P, w, p, sw, cw, nullptr, nullptr, nullptr, nullptr, out);
    }
}
